// HybridDensityQNN_63505386439247
// MI455X (gfx1250) — matrix operand order repaired and checked by reading, not compiled, not run on hardware
//
#include <hip/hip_runtime.h>
#include <hip/hip_bf16.h>
#include <math.h>

typedef __attribute__((ext_vector_type(16))) _Float16 v16h;
typedef __attribute__((ext_vector_type(8)))  _Float16 v8h;
typedef __attribute__((ext_vector_type(4)))  _Float16 v4h;
typedef __attribute__((ext_vector_type(2)))  _Float16 v2h;
typedef __attribute__((ext_vector_type(8)))  float    v8f;
typedef __attribute__((ext_vector_type(4)))  float    v4f;

// WMMA f16 fragment layouts (CDNA5 ISA §7.12.2, wave32):
//  A 16x32 (MxK): lane=(kh<<4)|m, element e -> k = (e>>3)*16 + kh*8 + (e&7)
//  B 32x16 (KxN): lane=(kh<<4)|n, element e -> same k as A; this file feeds A and B both k = kh*16 + e
//  C/D 16x16   : lane=(mh<<4)|n, vgpr r    -> M = r + 8*mh

// Pack 16 zero-extended u16 gathers into a v16h fragment (elem 2i = lo half).
union frag16 { unsigned u[8]; v16h h; };

__device__ __forceinline__ float fast_tanh(float x) {
  x = fminf(fmaxf(x, -10.f), 10.f);
  float e = __expf(2.f * x);
  return (e - 1.f) / (e + 1.f);
}

// Async global->LDS copy primitives (CDNA5 ISA, GV mode).
// LDS offset from the vdst VGPR, 64-bit global address from vaddr; tracked by ASYNCcnt.
__device__ __forceinline__ void async_g2l_b32(unsigned lds_off, const void* gaddr) {
  asm volatile("global_load_async_to_lds_b32 %0, %1, off"
               :: "v"(lds_off), "v"(gaddr) : "memory");
}
__device__ __forceinline__ void async_g2l_b128(unsigned lds_off, const void* gaddr) {
  asm volatile("global_load_async_to_lds_b128 %0, %1, off"
               :: "v"(lds_off), "v"(gaddr) : "memory");
}
__device__ __forceinline__ void wait_async0() {
  asm volatile("s_wait_asynccnt 0x0" ::: "memory");
}

// =====================================================================
// conv1 implicit GEMM, one block per image (784 = 49 exact M-tiles).
// =====================================================================
__global__ void __launch_bounds__(256)
conv1_gemm(const float* __restrict__ x,     // [B,3,32,32]
           const float* __restrict__ w1,    // [8,3,5,5]
           const float* __restrict__ b1,    // [8]
           _Float16* __restrict__ out,      // [B,8,28,28] f16
           int B) {
  __shared__ alignas(16) _Float16 sx[4000];      // 3072 image + zero pad
  __shared__ alignas(16) _Float16 swT[16 * 96];  // [n][k] padded
  __shared__ int kofs[96];

  const int tid = threadIdx.x;
  const int b   = blockIdx.x;

  const v4f* x4 = (const v4f*)(x + (size_t)b * 3072);
  for (int i = tid; i < 768; i += 256) {                 // image f32 -> f16
    v4f v = x4[i];
    v4h h; h.x = (_Float16)v.x; h.y = (_Float16)v.y;
    h.z = (_Float16)v.z; h.w = (_Float16)v.w;
    *(v4h*)(sx + i * 4) = h;
  }
  for (int i = tid; i < 232; i += 256) {                 // zero pad region
    v4h z = {};
    *(v4h*)(sx + 3072 + i * 4) = z;
  }
  for (int i = tid; i < 96; i += 256) {                  // im2col k -> offset
    int c = i / 25, r = i - c * 25, dy = r / 5, dx = r - dy * 5;
    kofs[i] = (i < 75) ? (c * 1024 + dy * 32 + dx) : 3072;
  }
  for (int i = tid; i < 16 * 96; i += 256) {             // weights [n][96]
    int n = i / 96, k = i - n * 96;
    swT[i] = (n < 8 && k < 75) ? (_Float16)w1[n * 75 + k] : (_Float16)0.f;
  }
  __syncthreads();

  const int lane = tid & 31, wv = tid >> 5;
  const int mrow = lane & 15, kh = lane >> 4, nn = lane & 15;

  // hoist: per-lane im2col offsets, B fragments, bias (all loop-invariant)
  int kreg[48];
#pragma unroll
  for (int kc = 0; kc < 3; ++kc)
#pragma unroll
    for (int e = 0; e < 16; ++e)
      kreg[kc * 16 + e] = kofs[kc * 32 + (kh << 4) + e];

  v16h bfrag[3];
#pragma unroll
  for (int kc = 0; kc < 3; ++kc)
    bfrag[kc] = *(const v16h*)(swT + nn * 96 + kc * 32 + kh * 16);

  const float bias = b1[nn & 7];
  _Float16* dst = out + (size_t)(b * 8 + (nn & 7)) * 784;

  for (int t = wv; t < 49; t += 8) {
    const int p = t * 16 + mrow;                 // < 784 always
    const int oy = p / 28, ox = p - oy * 28;
    const int base = oy * 32 + ox;

    // gather ALL 48 elements first -> loads batch behind few waits
    unsigned short tg[48];
#pragma unroll
    for (int i = 0; i < 48; ++i)
      tg[i] = *(const unsigned short*)(sx + base + kreg[i]);
    frag16 A[3];
#pragma unroll
    for (int kc = 0; kc < 3; ++kc)
#pragma unroll
      for (int j = 0; j < 8; ++j)
        A[kc].u[j] = (unsigned)tg[kc * 16 + 2 * j] |
                     ((unsigned)tg[kc * 16 + 2 * j + 1] << 16);

    v8f acc = {};
#pragma unroll
    for (int kc = 0; kc < 3; ++kc)
      acc = __builtin_amdgcn_wmma_f32_16x16x32_f16(false, A[kc].h, false, bfrag[kc],
                                                   (short)0, acc, false, false);

    if (nn < 8) {                                // packed b128 store, 8 rows
      v8h o;
#pragma unroll
      for (int r = 0; r < 8; ++r) o[r] = (_Float16)fmaxf(acc[r] + bias, 0.f);
      *(v8h*)(dst + t * 16 + 8 * kh) = o;
    }
  }
}

// =====================================================================
// 2x2 max-pool on f16 NCHW planes (paired v2h loads; strides all even)
// =====================================================================
__global__ void pool2x2_f16(const _Float16* __restrict__ in,
                            _Float16* __restrict__ out,
                            int total, int outW, int inW) {
  int i = blockIdx.x * blockDim.x + threadIdx.x;
  if (i >= total) return;
  int xo = i % outW; int t = i / outW;
  int yo = t % outW; int bc = t / outW;
  const _Float16* p = in + (size_t)bc * (inW * inW) + (2 * yo) * inW + 2 * xo;
  v2h a = *(const v2h*)p;
  v2h c = *(const v2h*)(p + inW);
  float m = fmaxf(fmaxf((float)a.x, (float)a.y), fmaxf((float)c.x, (float)c.y));
  out[i] = (_Float16)m;
}

// =====================================================================
// conv2 implicit GEMM, one block per image (100 -> 7 M-tiles, last masked).
// Input image staged via async global->LDS DMA (ASYNCcnt).
// =====================================================================
__global__ void __launch_bounds__(256)
conv2_gemm(const _Float16* __restrict__ in, // [B,8,14,14] f16
           const float* __restrict__ w2,    // [16,8,5,5]
           const float* __restrict__ b2,    // [16]
           _Float16* __restrict__ out,      // [B,16,10,10] f16
           int B) {
  __shared__ alignas(16) _Float16 sx[1728];       // 1568 + zero pad
  __shared__ alignas(16) _Float16 swT[16 * 224];  // [n][k] padded
  __shared__ int kofs[224];

  const int tid = threadIdx.x, b = blockIdx.x;

  { // async copy: pooled image -> LDS, no VGPR bounce
    const unsigned sxbase = (unsigned)(size_t)(&sx[0]);
    const unsigned* gsrc = (const unsigned*)(in + (size_t)b * 1568);
    for (int i = tid; i < 784; i += 256)
      async_g2l_b32(sxbase + (unsigned)i * 4u, gsrc + i);
  }
  for (int i = tid; i < 80; i += 256) ((unsigned*)(sx + 1568))[i] = 0u;
  for (int i = tid; i < 224; i += 256) {
    int c = i / 25, r = i - c * 25, dy = r / 5, dx = r - dy * 5;
    kofs[i] = (i < 200) ? (c * 196 + dy * 14 + dx) : 1568;
  }
  for (int i = tid; i < 16 * 224; i += 256) {
    int n = i / 224, k = i - n * 224;
    swT[i] = (k < 200) ? (_Float16)w2[n * 200 + k] : (_Float16)0.f;
  }
  wait_async0();
  __syncthreads();

  const int lane = tid & 31, wv = tid >> 5;
  const int mrow = lane & 15, kh = lane >> 4, nn = lane & 15;
  if (wv >= 7) return;                        // wave-uniform, after barrier

  const int p  = wv * 16 + mrow;
  const int pc = p < 100 ? p : 99;            // clamp padded rows (discarded)
  const int oy = pc / 10, ox = pc - oy * 10;
  const int base = oy * 14 + ox;
  v8f acc = {};
#pragma unroll
  for (int kc = 0; kc < 7; ++kc) {
    int kk[16];
#pragma unroll
    for (int e = 0; e < 16; ++e)
      kk[e] = kofs[kc * 32 + (kh << 4) + e];
    unsigned short tg[16];
#pragma unroll
    for (int e = 0; e < 16; ++e)
      tg[e] = *(const unsigned short*)(sx + base + kk[e]);
    frag16 A;
#pragma unroll
    for (int j = 0; j < 8; ++j)
      A.u[j] = (unsigned)tg[2 * j] | ((unsigned)tg[2 * j + 1] << 16);
    const v16h bfr = *(const v16h*)(swT + nn * 224 + kc * 32 + kh * 16);
    acc = __builtin_amdgcn_wmma_f32_16x16x32_f16(false, A.h, false, bfr,
                                                 (short)0, acc, false, false);
  }
  const float bias = b2[nn];
  _Float16* dst = out + (size_t)(b * 16 + nn) * 100;
  const int p0 = wv * 16 + 8 * kh;
  if (p0 + 8 <= 100) {                        // packed b64 stores
    v4h o0, o1;
#pragma unroll
    for (int r = 0; r < 4; ++r) {
      o0[r] = (_Float16)fmaxf(acc[r] + bias, 0.f);
      o1[r] = (_Float16)fmaxf(acc[4 + r] + bias, 0.f);
    }
    *(v4h*)(dst + p0) = o0;
    *(v4h*)(dst + p0 + 4) = o1;
  } else {
#pragma unroll
    for (int r = 0; r < 8; ++r) {
      const int pr = p0 + r;
      if (pr < 100) dst[pr] = (_Float16)fmaxf(acc[r] + bias, 0.f);
    }
  }
}

// =====================================================================
// fc1: angles[B,4] = tanh(feat[B,400] @ w[4,400]^T + b), 128 rows / block.
// Activation tile staged via async global->LDS b128 DMA.
// =====================================================================
__global__ void __launch_bounds__(256)
fc1_gemm(const _Float16* __restrict__ feat, // [B,400] f16
         const float* __restrict__ w,       // [4,400]
         const float* __restrict__ bia,     // [4]
         float* __restrict__ angles,        // [B,4] f32
         int B) {
  __shared__ alignas(16) _Float16 sfeat[128 * 400];   // 100 KB
  __shared__ alignas(16) _Float16 swT[16 * 416];      // [n][k] padded

  const int tid = threadIdx.x;
  const int g0  = blockIdx.x * 128;
  const int rows = (B - g0) < 128 ? (B - g0) : 128;

  { // async tile copy, 16B granules
    const unsigned sbase = (unsigned)(size_t)(&sfeat[0]);
    const uint4* src = (const uint4*)(feat + (size_t)g0 * 400);
    const int n4 = rows * 50;                          // 400 f16 = 50 uint4
    for (int i = tid; i < n4; i += 256)
      async_g2l_b128(sbase + (unsigned)i * 16u, src + i);
  }
  for (int i = tid; i < 16 * 416; i += 256) {
    int n = i / 416, k = i - n * 416;
    swT[i] = (n < 4 && k < 400) ? (_Float16)w[n * 400 + k] : (_Float16)0.f;
  }
  wait_async0();
  __syncthreads();

  const int lane = tid & 31, wv = tid >> 5;
  const int mrow = lane & 15, kh = lane >> 4, nn = lane & 15;

  const _Float16* arow = sfeat + (size_t)(wv * 16 + mrow) * 400;
  const float bias = bia[nn & 3];
  v8f acc = {};
#pragma unroll
  for (int kc = 0; kc < 12; ++kc) {                    // k = 0..383: no masks
    v8h lo = *(const v8h*)(arow + kc * 32 + kh * 16);
    v8h hi = *(const v8h*)(arow + kc * 32 + kh * 16 + 8);
    v16h afr = __builtin_shufflevector(lo, hi, 0, 1, 2, 3, 4, 5, 6, 7,
                                       8, 9, 10, 11, 12, 13, 14, 15);
    const v16h bfr = *(const v16h*)(swT + nn * 416 + kc * 32 + kh * 16);
    acc = __builtin_amdgcn_wmma_f32_16x16x32_f16(false, afr, false, bfr,
                                                 (short)0, acc, false, false);
  }
  { // tail chunk k = 384..415, valid k < 400
    unsigned short tg[16];
#pragma unroll
    for (int e = 0; e < 16; ++e) {
      const int ka = 384 + (kh << 4) + e;
      tg[e] = (ka < 400) ? *(const unsigned short*)(arow + ka) : (unsigned short)0;
    }
    frag16 A;
#pragma unroll
    for (int j = 0; j < 8; ++j)
      A.u[j] = (unsigned)tg[2 * j] | ((unsigned)tg[2 * j + 1] << 16);
    const v16h bfr = *(const v16h*)(swT + nn * 416 + 384 + kh * 16);
    acc = __builtin_amdgcn_wmma_f32_16x16x32_f16(false, A.h, false, bfr,
                                                 (short)0, acc, false, false);
  }
  if (nn < 4) {
#pragma unroll
    for (int r = 0; r < 8; ++r) {
      const int g = g0 + wv * 16 + r + 8 * kh;
      if (g < B) angles[(size_t)g * 4 + nn] = fast_tanh(acc[r] + bias);
    }
  }
}

// =====================================================================
// 4-qubit statevector circuit (16 complex amps in registers), K=2 weight
// sets, softmax(alpha) mix, then fc2 (4->10). One thread per batch element.
// =====================================================================
__global__ void quantum_fc2(const float* __restrict__ angles, // [B,4]
                            const float* __restrict__ qw,     // [2,8]
                            const float* __restrict__ alpha,  // [2]
                            const float* __restrict__ f2w,    // [10,4]
                            const float* __restrict__ f2b,    // [10]
                            float* __restrict__ out,          // [B,10]
                            int B) {
  int b = blockIdx.x * blockDim.x + threadIdx.x;
  if (b >= B) return;

  float a[4];
#pragma unroll
  for (int i = 0; i < 4; ++i) a[i] = angles[(size_t)b * 4 + i];

  float al0 = alpha[0], al1 = alpha[1];
  float mx = fmaxf(al0, al1);
  float e0 = __expf(al0 - mx), e1 = __expf(al1 - mx);
  float inv = 1.f / (e0 + e1);
  float wk[2] = { e0 * inv, e1 * inv };

  float q[4] = {0.f, 0.f, 0.f, 0.f};

#pragma unroll
  for (int kk = 0; kk < 2; ++kk) {
    float w[8];
#pragma unroll
    for (int j = 0; j < 8; ++j) w[j] = qw[kk * 8 + j];

    float re[16], im[16];
#pragma unroll
    for (int s = 0; s < 16; ++s) { re[s] = 0.f; im[s] = 0.f; }
    re[0] = 1.f;

    auto ry = [&](int wire, float theta) {
      float s, c;
      __sincosf(0.5f * theta, &s, &c);
      int bit = 8 >> wire;
#pragma unroll
      for (int idx = 0; idx < 16; ++idx)
        if (!(idx & bit)) {
          int j = idx | bit;
          float r0 = re[idx], i0 = im[idx], r1 = re[j], i1 = im[j];
          re[idx] = c * r0 - s * r1;  im[idx] = c * i0 - s * i1;
          re[j]   = s * r0 + c * r1;  im[j]   = s * i0 + c * i1;
        }
    };

#pragma unroll
    for (int i = 0; i < 4; ++i) ry(i, a[i]);            // RY(a_i)

#pragma unroll
    for (int i = 0; i < 4; ++i) {                       // RZ(w_i)
      float s, c;
      __sincosf(0.5f * w[i], &s, &c);
      int bit = 8 >> i;
#pragma unroll
      for (int idx = 0; idx < 16; ++idx) {
        float sr = (idx & bit) ? s : -s;
        float r = re[idx], ii = im[idx];
        re[idx] = c * r - sr * ii;
        im[idx] = c * ii + sr * r;
      }
    }

#pragma unroll
    for (int ci = 0; ci < 3; ++ci) {                    // CNOT chain
      int cb = 8 >> ci, tb = 8 >> (ci + 1);
#pragma unroll
      for (int idx = 0; idx < 16; ++idx)
        if ((idx & cb) && !(idx & tb)) {
          int j = idx | tb;
          float tr = re[idx]; re[idx] = re[j]; re[j] = tr;
          float ti = im[idx]; im[idx] = im[j]; im[j] = ti;
        }
    }

#pragma unroll
    for (int i = 0; i < 4; ++i) ry(i, w[4 + i]);        // RY(w[4+i])

    float p[16];
#pragma unroll
    for (int s = 0; s < 16; ++s) p[s] = re[s] * re[s] + im[s] * im[s];
#pragma unroll
    for (int i = 0; i < 4; ++i) {
      int bit = 8 >> i;
      float ez = 0.f;
#pragma unroll
      for (int s = 0; s < 16; ++s) ez += (s & bit) ? -p[s] : p[s];
      q[i] += wk[kk] * ez;
    }
  }

#pragma unroll
  for (int o = 0; o < 10; ++o) {
    float s = f2b[o];
#pragma unroll
    for (int n = 0; n < 4; ++n) s += q[n] * f2w[o * 4 + n];
    out[(size_t)b * 10 + o] = s;
  }
}

// =====================================================================
extern "C" void kernel_launch(void* const* d_in, const int* in_sizes, int n_in,
                              void* d_out, int out_size, void* d_ws, size_t ws_size,
                              hipStream_t stream) {
  const float* x   = (const float*)d_in[0];
  const float* c1w = (const float*)d_in[1];
  const float* c1b = (const float*)d_in[2];
  const float* c2w = (const float*)d_in[3];
  const float* c2b = (const float*)d_in[4];
  const float* f1w = (const float*)d_in[5];
  const float* f1b = (const float*)d_in[6];
  const float* qw  = (const float*)d_in[7];
  const float* alp = (const float*)d_in[8];
  const float* f2w = (const float*)d_in[9];
  const float* f2b = (const float*)d_in[10];
  float* out = (float*)d_out;
  (void)n_in; (void)out_size; (void)ws_size;

  const int B = in_sizes[0] / (3 * 32 * 32);

  char* ws = (char*)d_ws;
  size_t off = 0;
  auto take = [&](size_t bytes) {
    char* p = ws + off;
    off = (off + bytes + 255) & ~(size_t)255;
    return p;
  };
  _Float16* conv1o = (_Float16*)take((size_t)B * 8 * 784 * sizeof(_Float16));  // [B,8,28,28]
  _Float16* pool1o = (_Float16*)take((size_t)B * 8 * 196 * sizeof(_Float16));  // [B,8,14,14]
  _Float16* conv2o = (_Float16*)take((size_t)B * 16 * 100 * sizeof(_Float16)); // [B,16,10,10]
  _Float16* feat   = (_Float16*)take((size_t)B * 400 * sizeof(_Float16));      // [B,400]
  float*    ang    = (float*)   take((size_t)B * 4 * sizeof(float));           // [B,4]

  conv1_gemm<<<B, 256, 0, stream>>>(x, c1w, c1b, conv1o, B);
  {
    int total = B * 8 * 196;
    pool2x2_f16<<<(total + 255) / 256, 256, 0, stream>>>(conv1o, pool1o, total, 14, 28);
  }
  conv2_gemm<<<B, 256, 0, stream>>>(pool1o, c2w, c2b, conv2o, B);
  {
    int total = B * 400;
    pool2x2_f16<<<(total + 255) / 256, 256, 0, stream>>>(conv2o, feat, total, 5, 10);
  }
  fc1_gemm<<<(B + 127) / 128, 256, 0, stream>>>(feat, f1w, f1b, ang, B);
  quantum_fc2<<<(B + 255) / 256, 256, 0, stream>>>(ang, qw, alp, f2w, f2b, out, B);
}
